// GraphConvEncoder_50268297232879
// MI455X (gfx1250) — matrix operand order repaired and checked by reading, not compiled, not run on hardware
//
#include <hip/hip_runtime.h>
#include <stddef.h>

#define HDIM 256
#define NGRAPHS 256
#define NEGV (-1e9f)

typedef __attribute__((ext_vector_type(16))) __bf16 v16bf;
typedef __attribute__((ext_vector_type(8)))  float v8f;

#if __has_builtin(__builtin_amdgcn_global_load_async_to_lds_b128) && \
    __has_builtin(__builtin_amdgcn_s_wait_asynccnt)
#define USE_ASYNC 1
typedef __attribute__((ext_vector_type(4))) int v4i;
typedef __attribute__((address_space(1))) v4i* gv4i_p;   // global int4*
typedef __attribute__((address_space(3))) v4i* lv4i_p;   // LDS int4*
#else
#define USE_ASYNC 0
#endif

__device__ __forceinline__ unsigned short f2bf(float f) {
  unsigned u = __builtin_bit_cast(unsigned, f);
  u += 0x7FFFu + ((u >> 16) & 1u);           // round-to-nearest-even
  return (unsigned short)(u >> 16);
}
__device__ __forceinline__ float bf2f(unsigned short h) {
  unsigned u = ((unsigned)h) << 16;
  return __builtin_bit_cast(float, u);
}

union FragBF { v16bf v; unsigned u[8]; };

// ---- W pre-convert: f32 [256,256] -> bf16 hi/lo, tile-contiguous layout ----
// dst tile order: [nb(4)][ktI(8)] each tile = [nn(64)][kk(32)] shorts (4KB)
__global__ void k_wconv(const float* __restrict__ W, unsigned short* __restrict__ whi,
                        unsigned short* __restrict__ wlo) {
  int i = blockIdx.x * 256 + threadIdx.x;            // i = k*256 + n
  int k = i >> 8, n = i & 255;
  float v = W[i];
  int dst = (((n >> 6) << 3) + (k >> 5)) * 2048 + (n & 63) * 32 + (k & 31);
  unsigned short hi = f2bf(v);
  whi[dst] = hi;
  wlo[dst] = f2bf(v - bf2f(hi));
}

// ---------------- GEMM: Dst[M,256] = A[M,256] @ W[256,256], bf16x3 split ----
__global__ __launch_bounds__(256) void k_gemm_bf16x3(
    const float* __restrict__ A,
    const unsigned short* __restrict__ whiG, const unsigned short* __restrict__ wloG,
    float* __restrict__ Dst, int M)
{
  __shared__ unsigned short Ahi[64*34], Alo[64*34];
  __shared__ __align__(16) unsigned short Bhi[64*32], Blo[64*32];
  const int mBase = (blockIdx.x >> 2) * 64;
  const int nBase = (blockIdx.x & 3) * 64;
  const int tid  = threadIdx.x;
  const int lane = tid & 31, wave = tid >> 5;
  const int wm = wave & 3, wn = wave >> 2;
  const int half = lane >> 4;
  const bool full = (mBase + 64 <= M);
  v8f acc0 = {};
  v8f acc1 = {};

  for (int kt = 0; kt < HDIM; kt += 32) {
    // ---- B tiles: async copy pre-converted bf16 straight into LDS ----------
    const size_t tOff = (size_t)(((nBase >> 6) << 3) + (kt >> 5)) * 2048 + tid * 8;
#if USE_ASYNC
    __builtin_amdgcn_global_load_async_to_lds_b128((gv4i_p)(whiG + tOff),
                                                   (lv4i_p)(Bhi + tid * 8), 0, 0);
    __builtin_amdgcn_global_load_async_to_lds_b128((gv4i_p)(wloG + tOff),
                                                   (lv4i_p)(Blo + tid * 8), 0, 0);
#else
    *(uint4*)(Bhi + tid * 8) = *(const uint4*)(whiG + tOff);
    *(uint4*)(Blo + tid * 8) = *(const uint4*)(wloG + tOff);
#endif
    // ---- A tile 64x32: load f32, split to bf16 hi/lo (overlaps async B) ----
    if (full) {
#pragma unroll
      for (int it = 0; it < 8; ++it) {
        int i = tid + it * 256;
        int r = i >> 5, c = i & 31;
        float v = A[(size_t)(mBase + r) * HDIM + kt + c];
        unsigned short hi = f2bf(v);
        Ahi[r*34 + c] = hi;
        Alo[r*34 + c] = f2bf(v - bf2f(hi));
      }
      if (kt + 32 < HDIM)
        __builtin_prefetch(&A[(size_t)(mBase + (tid >> 2)) * HDIM + kt + 32], 0, 1);
    } else {
#pragma unroll
      for (int it = 0; it < 8; ++it) {
        int i = tid + it * 256;
        int r = i >> 5, c = i & 31;
        int gr = mBase + r;
        float v = (gr < M) ? A[(size_t)gr * HDIM + kt + c] : 0.0f;
        unsigned short hi = f2bf(v);
        Ahi[r*34 + c] = hi;
        Alo[r*34 + c] = f2bf(v - bf2f(hi));
      }
    }
#if USE_ASYNC
    __builtin_amdgcn_s_wait_asynccnt(0);
#endif
    __syncthreads();

    FragBF fAh, fAl, fB0h, fB0l, fB1h, fB1l;
    {
      const int arow = wm * 16 + (lane & 15);
      const unsigned* rh = (const unsigned*)(Ahi + arow * 34);
      const unsigned* rl = (const unsigned*)(Alo + arow * 34);
#pragma unroll
      for (int v = 0; v < 8; ++v) {
        int p = ((v < 4) ? 0 : 8) + half * 4 + (v & 3);   // ISA A 16-bit 16x32 layout
        fAh.u[v] = rh[p]; fAl.u[v] = rl[p];
      }
    }
    {
      const int n0 = wn * 32 + (lane & 15);
      const unsigned* b0h = (const unsigned*)(Bhi + n0 * 32);
      const unsigned* b0l = (const unsigned*)(Blo + n0 * 32);
      const unsigned* b1h = (const unsigned*)(Bhi + (n0 + 16) * 32);
      const unsigned* b1l = (const unsigned*)(Blo + (n0 + 16) * 32);
#pragma unroll
      for (int v = 0; v < 8; ++v) {
        int p = ((v < 4) ? 0 : 8) + half * 4 + (v & 3);   // ISA B 16-bit 32x16 layout
        fB0h.u[v] = b0h[p]; fB0l.u[v] = b0l[p];
        fB1h.u[v] = b1h[p]; fB1l.u[v] = b1l[p];
      }
    }
    // D += Ahi*Bhi + Alo*Bhi + Ahi*Blo  (bf16x3 ~ f32 accuracy)
    acc0 = __builtin_amdgcn_wmma_f32_16x16x32_bf16(false, fAh.v, false, fB0h.v, (short)0, acc0, false, false);
    acc0 = __builtin_amdgcn_wmma_f32_16x16x32_bf16(false, fAl.v, false, fB0h.v, (short)0, acc0, false, false);
    acc0 = __builtin_amdgcn_wmma_f32_16x16x32_bf16(false, fAh.v, false, fB0l.v, (short)0, acc0, false, false);
    acc1 = __builtin_amdgcn_wmma_f32_16x16x32_bf16(false, fAh.v, false, fB1h.v, (short)0, acc1, false, false);
    acc1 = __builtin_amdgcn_wmma_f32_16x16x32_bf16(false, fAl.v, false, fB1h.v, (short)0, acc1, false, false);
    acc1 = __builtin_amdgcn_wmma_f32_16x16x32_bf16(false, fAh.v, false, fB1l.v, (short)0, acc1, false, false);
    __syncthreads();
  }
  const int col   = nBase + wn * 32 + (lane & 15);
  const int rbase = mBase + wm * 16 + half * 8;
  if (full) {
#pragma unroll
    for (int i = 0; i < 8; ++i) {
      Dst[(size_t)(rbase + i) * HDIM + col]      = acc0[i];
      Dst[(size_t)(rbase + i) * HDIM + col + 16] = acc1[i];
    }
  } else {
#pragma unroll
    for (int i = 0; i < 8; ++i) {
      int gr = rbase + i;
      if (gr < M) {
        Dst[(size_t)gr * HDIM + col]      = acc0[i];
        Dst[(size_t)gr * HDIM + col + 16] = acc1[i];
      }
    }
  }
}

// ---------------- utility kernels --------------------------------------------
__global__ void k_fill_f32(float* p, float v, long n) {
  long i = (long)blockIdx.x * blockDim.x + threadIdx.x;
  if (i < n) p[i] = v;
}
__global__ void k_fill_i32(int* p, int v, long n) {
  long i = (long)blockIdx.x * blockDim.x + threadIdx.x;
  if (i < n) p[i] = v;
}
__global__ void k_scal(float* s) {
  if (threadIdx.x == 0) { s[1] = 1.0f; s[2] = 0.0f; }
}
__global__ void k_hist(const int* __restrict__ batch, int* __restrict__ counts, int M) {
  int i = blockIdx.x * 256 + threadIdx.x;
  if (i < M) atomicAdd(&counts[batch[i]], 1);
}
__global__ void k_starts(const int* __restrict__ counts, int* __restrict__ starts) {
  __shared__ int sh[NGRAPHS];
  int t = threadIdx.x;
  sh[t] = counts[t]; __syncthreads();
  for (int off = 1; off < NGRAPHS; off <<= 1) {
    int v = (t >= off) ? sh[t - off] : 0;
    __syncthreads();
    sh[t] += v;
    __syncthreads();
  }
  starts[t] = sh[t] - counts[t];
}

// ---------------- GCN aggregation --------------------------------------------
__global__ void k_deg(const int* __restrict__ dst, const int* __restrict__ ealive,
                      float* __restrict__ deg, int E) {
  int i = blockIdx.x * 256 + threadIdx.x;
  if (i < E && ealive[i]) atomicAdd(&deg[dst[i]], 1.0f);
}
__global__ void k_dis(float* deg, int M) {
  int i = blockIdx.x * 256 + threadIdx.x;
  if (i < M) deg[i] = rsqrtf(deg[i] + 1.0f);     // deg buffer becomes dis
}
__global__ void k_coef(const int* __restrict__ src, const int* __restrict__ dst,
                       const int* __restrict__ ealive, const float* __restrict__ dis,
                       float* __restrict__ coef, int E) {
  int i = blockIdx.x * 256 + threadIdx.x;
  if (i < E) coef[i] = ealive[i] ? dis[src[i]] * dis[dst[i]] : 0.0f;
}
__global__ void k_agginit(const float* __restrict__ hW, const float* __restrict__ dis,
                          const float* __restrict__ b, float* __restrict__ agg, int M) {
  long i = (long)blockIdx.x * blockDim.x + threadIdx.x;
  if (i < (long)M * HDIM) {
    int node = (int)(i >> 8), f = (int)(i & 255);
    float d = dis[node];
    agg[i] = hW[i] * d * d + b[f];               // self-loop message + bias
  }
}
__global__ void k_scatter(const float* __restrict__ hW, const int* __restrict__ src,
                          const int* __restrict__ dst, const float* __restrict__ coef,
                          float* __restrict__ agg, int E) {
  int e = blockIdx.x;
  float c = coef[e];
  if (c == 0.0f) return;
  int s = src[e], d = dst[e], f = threadIdx.x;
  atomicAdd(&agg[(size_t)d * HDIM + f], hW[(size_t)s * HDIM + f] * c);
}

// ---------------- fused pooling passes ---------------------------------------
// h = relu(h) in place; score = alive ? dot(h,p)*scale : NEG   (wave per node)
__global__ void k_relu_dot(float* __restrict__ h, const float* __restrict__ vec,
                           const float* __restrict__ scalep, const int* __restrict__ alive,
                           float* __restrict__ score, int M) {
  int node = blockIdx.x * 8 + (threadIdx.x >> 5);
  int lane = threadIdx.x & 31;
  if (node >= M) return;
  float* row = h + (size_t)node * HDIM;
  float s = 0.0f;
#pragma unroll
  for (int f = 0; f < 8; ++f) {
    int c = lane + f * 32;
    float v = fmaxf(row[c], 0.0f);
    row[c] = v;
    s += v * vec[c];
  }
  for (int off = 16; off; off >>= 1) s += __shfl_xor(s, off, 32);
  if (lane == 0) score[node] = alive[node] ? s * scalep[0] : NEGV;
}
// h *= mult[node] in place; gate = alive ? dot(h,gw)+gb : NEG
__global__ void k_scale_dot(float* __restrict__ h, const float* __restrict__ mult,
                            const float* __restrict__ gw, const float* __restrict__ gb,
                            const int* __restrict__ alive, float* __restrict__ gate, int M) {
  int node = blockIdx.x * 8 + (threadIdx.x >> 5);
  int lane = threadIdx.x & 31;
  if (node >= M) return;
  float* row = h + (size_t)node * HDIM;
  float mu = mult[node];
  float s = 0.0f;
#pragma unroll
  for (int f = 0; f < 8; ++f) {
    int c = lane + f * 32;
    float v = row[c] * mu;
    row[c] = v;
    s += v * gw[c];
  }
  for (int off = 16; off; off >>= 1) s += __shfl_xor(s, off, 32);
  if (lane == 0) gate[node] = alive[node] ? (s + gb[0]) : NEGV;
}

__global__ void k_pnorm(const float* __restrict__ p, float* __restrict__ out) {
  __shared__ float sh[HDIM];
  int t = threadIdx.x;
  float v = p[t];
  sh[t] = v * v; __syncthreads();
  for (int s = 128; s; s >>= 1) { if (t < s) sh[t] += sh[t + s]; __syncthreads(); }
  if (t == 0) out[0] = rsqrtf(sh[0]);            // 1/||p||
}
__global__ __launch_bounds__(256) void k_topk(const float* __restrict__ score,
    const int* __restrict__ starts, const int* __restrict__ counts,
    int* __restrict__ alive, float* __restrict__ mult) {
  int g = blockIdx.x, t = threadIdx.x;
  int n0 = starts[g], cnt = counts[g];
  __shared__ int red[256];
  int loc = 0;
  for (int i = t; i < cnt; i += 256) loc += alive[n0 + i];
  red[t] = loc; __syncthreads();
  for (int s = 128; s; s >>= 1) { if (t < s) red[t] += red[t + s]; __syncthreads(); }
  int k = (int)ceilf(0.8f * (float)red[0]);
  for (int i = t; i < cnt; i += 256) {
    int idx = n0 + i;
    float si = score[idx];
    int rank = 0;
    for (int j = 0; j < cnt; ++j) {             // stable rank, ties by index
      float sj = score[n0 + j];
      rank += (sj > si) || (sj == si && j < i);
    }
    int keep = (alive[idx] != 0) && (rank < k);
    mult[idx] = keep ? tanhf(si) : 0.0f;
    alive[idx] = keep;
  }
}
__global__ void k_eupd(int* __restrict__ ealive, const int* __restrict__ src,
                       const int* __restrict__ dst, const int* __restrict__ alive, int E) {
  int i = blockIdx.x * 256 + threadIdx.x;
  if (i < E) ealive[i] = ealive[i] && alive[src[i]] && alive[dst[i]];
}
__global__ __launch_bounds__(256) void k_attn(const float* __restrict__ h,
    const float* __restrict__ gate, const int* __restrict__ alive,
    const int* __restrict__ starts, const int* __restrict__ counts,
    float* __restrict__ out) {
  int g = blockIdx.x, t = threadIdx.x;
  int n0 = starts[g], cnt = counts[g];
  __shared__ float red[256];
  float m = -3e38f;
  for (int i = t; i < cnt; i += 256) m = fmaxf(m, gate[n0 + i]);
  red[t] = m; __syncthreads();
  for (int s = 128; s; s >>= 1) { if (t < s) red[t] = fmaxf(red[t], red[t + s]); __syncthreads(); }
  m = red[0]; __syncthreads();
  float z = 0.0f;
  for (int i = t; i < cnt; i += 256) {
    int idx = n0 + i;
    if (alive[idx]) z += expf(gate[idx] - m);
  }
  red[t] = z; __syncthreads();
  for (int s = 128; s; s >>= 1) { if (t < s) red[t] += red[t + s]; __syncthreads(); }
  z = fmaxf(red[0], 1e-12f);
  float acc = 0.0f;
  for (int i = 0; i < cnt; ++i) {
    int idx = n0 + i;
    if (alive[idx]) acc += expf(gate[idx] - m) * h[(size_t)idx * HDIM + t];
  }
  out[g * HDIM + t] += acc / z;
}

// ---------------- driver -----------------------------------------------------
extern "C" void kernel_launch(void* const* d_in, const int* in_sizes, int n_in,
                              void* d_out, int out_size, void* d_ws, size_t ws_size,
                              hipStream_t stream) {
  const float* x     = (const float*)d_in[0];
  const int*   ei    = (const int*)d_in[1];
  const int*   batch = (const int*)d_in[2];
  const float* Ws[3] = {(const float*)d_in[3], (const float*)d_in[6], (const float*)d_in[9]};
  const float* bs[3] = {(const float*)d_in[4], (const float*)d_in[7], (const float*)d_in[10]};
  const float* ps[3] = {(const float*)d_in[5], (const float*)d_in[8], (const float*)d_in[11]};
  const float* gw = (const float*)d_in[12];
  const float* gb = (const float*)d_in[13];
  const int M = in_sizes[0] / HDIM;
  const int E = in_sizes[1] / 2;
  const int* src = ei;
  const int* dst = ei + E;

  char* w = (char*)d_ws;
  size_t off = 0;
  auto take = [&](size_t bytes) {
    void* p = w + off;
    off += (bytes + 255) & ~(size_t)255;
    return p;
  };
  float* bufA = (float*)take((size_t)M * HDIM * 4);
  float* bufB = (float*)take((size_t)M * HDIM * 4);
  float* bufC = (float*)take((size_t)M * HDIM * 4);
  float* dis  = (float*)take((size_t)M * 4);
  float* coef = (float*)take((size_t)E * 4);
  float* score= (float*)take((size_t)M * 4);
  float* mult = (float*)take((size_t)M * 4);
  float* scal = (float*)take(4 * 4);             // [0]=1/||p||, [1]=1.0, [2]=0.0
  int* alive  = (int*)take((size_t)M * 4);
  int* ealive = (int*)take((size_t)E * 4);
  int* counts = (int*)take(NGRAPHS * 4);
  int* starts = (int*)take(NGRAPHS * 4);
  unsigned short* whi = (unsigned short*)take((size_t)HDIM * HDIM * 2);
  unsigned short* wlo = (unsigned short*)take((size_t)HDIM * HDIM * 2);

  dim3 blk(256);
  auto gsz = [&](long n) { return dim3((unsigned)((n + 255) / 256)); };

  k_fill_f32<<<gsz(out_size), blk, 0, stream>>>((float*)d_out, 0.0f, (long)out_size);
  k_fill_i32<<<gsz(M), blk, 0, stream>>>(alive, 1, (long)M);
  k_fill_i32<<<gsz(E), blk, 0, stream>>>(ealive, 1, (long)E);
  k_fill_i32<<<gsz(NGRAPHS), blk, 0, stream>>>(counts, 0, (long)NGRAPHS);
  k_scal<<<1, 32, 0, stream>>>(scal);
  k_hist<<<gsz(M), blk, 0, stream>>>(batch, counts, M);
  k_starts<<<1, NGRAPHS, 0, stream>>>(counts, starts);

  const float* hin = x;
  float* aggs[3] = {bufB, bufC, bufB};
  float* out = (float*)d_out;
  const int mtiles = (M + 63) / 64;

  for (int l = 0; l < 3; ++l) {
    float* hW  = bufA;
    float* agg = aggs[l];
    // h = relu(GCN(hin))
    k_wconv<<<dim3(HDIM), blk, 0, stream>>>(Ws[l], whi, wlo);
    k_gemm_bf16x3<<<dim3(mtiles * 4), blk, 0, stream>>>(hin, whi, wlo, hW, M);
    k_fill_f32<<<gsz(M), blk, 0, stream>>>(dis, 0.0f, (long)M);
    k_deg<<<gsz(E), blk, 0, stream>>>(dst, ealive, dis, E);
    k_dis<<<gsz(M), blk, 0, stream>>>(dis, M);
    k_coef<<<gsz(E), blk, 0, stream>>>(src, dst, ealive, dis, coef, E);
    k_agginit<<<gsz((long)M * HDIM), blk, 0, stream>>>(hW, dis, bs[l], agg, M);
    k_scatter<<<dim3((unsigned)E), blk, 0, stream>>>(hW, src, dst, coef, agg, E);
    // TopK pooling (relu fused with score dot)
    k_pnorm<<<1, HDIM, 0, stream>>>(ps[l], scal);
    k_relu_dot<<<dim3((M + 7) / 8), blk, 0, stream>>>(agg, ps[l], scal, alive, score, M);
    k_topk<<<dim3(NGRAPHS), blk, 0, stream>>>(score, starts, counts, alive, mult);
    // scale fused with attention-gate dot
    k_scale_dot<<<dim3((M + 7) / 8), blk, 0, stream>>>(agg, mult, gw, gb, alive, score, M);
    k_eupd<<<gsz(E), blk, 0, stream>>>(ealive, src, dst, alive, E);
    k_attn<<<dim3(NGRAPHS), blk, 0, stream>>>(agg, score, alive, starts, counts, out);
    hin = agg;
  }
}
